// LSTMModel_34394098106639
// MI455X (gfx1250) — compile-verified, matrix operand order checked by reading, not run on hardware
//
#include <hip/hip_runtime.h>
#include <hip/hip_bf16.h>

typedef __attribute__((ext_vector_type(16))) _Float16 v16h;
typedef __attribute__((ext_vector_type(8)))  _Float16 v8h;
typedef __attribute__((ext_vector_type(8)))  float    v8f;

#define TSTEPS 120
#define BATCH  4096
#define D_IN   132

// ---------------------------------------------------------------------------
// Pack an f32 weight matrix W[K][N] into WMMA-fragment-major f16:
//   P[((kt*(N/16)+nt)*32 + lane)*16 + j]
// so one wave loads a 32x16 f16 B-fragment with a single 32B/lane load.
// Rows k >= K are zero-padded (layer-1 K=132 -> Kpad=160).
// ---------------------------------------------------------------------------
__global__ __launch_bounds__(256) void pack_weight_kernel(
    const float* __restrict__ W, _Float16* __restrict__ P, int K, int Kpad, int N)
{
    int idx = blockIdx.x * 256 + threadIdx.x;
    int total = Kpad * N;
    if (idx >= total) return;
    int j    = idx & 15;
    int lane = (idx >> 4) & 31;
    int tile = idx >> 9;             // kt*(N/16) + nt
    int NT   = N >> 4;
    int kt   = tile / NT;
    int nt   = tile % NT;
    int base = (lane >> 4) << 3;
    int kk   = base + (j < 8 ? j : j + 8);
    int k    = kt * 32 + kk;
    int n    = nt * 16 + (lane & 15);
    P[idx] = (k < K) ? (_Float16)W[(long long)k * N + n] : (_Float16)0.0f;
}

// ---------------------------------------------------------------------------
// GEMM: C[M][N] (f16) = A[M][K] @ Bp + bias.  Block tile 64x128, 8 waves;
// wave w owns m-tile w>>1 and four n-tiles (acc = 4 x v8f). 64xKpad A panel
// staged once in LDS as f16 (padded pitch -> conflict-free ds_load_b128).
// Layer-1 [B,T,K] -> [T,B,K] row permutation folded into the staging loader.
// ---------------------------------------------------------------------------
template <typename AT>
__global__ __launch_bounds__(256) void gemm_wmma_kernel(
    const AT* __restrict__ A, const _Float16* __restrict__ Bp,
    const float* __restrict__ bias, _Float16* __restrict__ C,
    int M, int N, int K, int Kpad, int perm, int PB, int PT)
{
    __shared__ _Float16 a_sh[64 * 168];   // max Kpad=160 (+8 pad pitch)
    const int pitch = Kpad + 8;

    const int w    = threadIdx.x >> 5;
    const int lane = threadIdx.x & 31;
    const int l15  = lane & 15;
    const int hb   = lane >> 4;
    const int base = hb << 3;

    const int mt     = w >> 1;
    const int ntBase = (w & 1) * 4;
    const long long m0 = (long long)blockIdx.x * 64;
    const int n0 = blockIdx.y * 128;

    // ---- stage A panel (convert to f16, zero-pad K) ----
    const int chunks = Kpad >> 3;
    for (int idx = threadIdx.x; idx < 64 * chunks; idx += 256) {
        int row = idx / chunks;
        int cb  = (idx % chunks) * 8;
        long long r = m0 + row;
        long long srow = perm ? ((long long)((int)r % PB) * PT + (int)(r / PB)) : r;
        const AT* ap = A + srow * K;
        _Float16* dp = &a_sh[row * pitch + cb];
#pragma unroll
        for (int j = 0; j < 8; ++j) {
            int col = cb + j;
            dp[j] = (col < K) ? (_Float16)ap[col] : (_Float16)0.0f;
        }
    }
    __syncthreads();

    // ---- accumulators seeded with bias ----
    v8f acc[4];
#pragma unroll
    for (int i = 0; i < 4; ++i) {
        float bv = bias[n0 + (ntBase + i) * 16 + l15];
#pragma unroll
        for (int j = 0; j < 8; ++j) acc[i][j] = bv;
    }

    const int KT = Kpad >> 5;
    const int NTtot = N >> 4;
    for (int kt = 0; kt < KT; ++kt) {
        const _Float16* rp = &a_sh[(mt * 16 + l15) * pitch + kt * 32 + base];
        v8h lo = *(const v8h*)rp;
        v8h hi = *(const v8h*)(rp + 16);
        v16h a;
#pragma unroll
        for (int j = 0; j < 8; ++j) { a[j] = lo[j]; a[j + 8] = hi[j]; }
#pragma unroll
        for (int i = 0; i < 4; ++i) {
            int ntG = (n0 >> 4) + ntBase + i;
            v16h b = *(const v16h*)(Bp + (((long long)kt * NTtot + ntG) * 32 + lane) * 16);
            acc[i] = __builtin_amdgcn_wmma_f32_16x16x32_f16(
                false, a, false, b, (short)0, acc[i], false, false);
        }
    }

    // ---- store C (f16) ----
#pragma unroll
    for (int i = 0; i < 4; ++i) {
        int n = n0 + (ntBase + i) * 16 + l15;
#pragma unroll
        for (int vr = 0; vr < 8; ++vr) {
            long long m = m0 + mt * 16 + vr + base;
            C[m * N + n] = (_Float16)acc[i][vr];
        }
    }
}

// ---------------------------------------------------------------------------
// Recurrent scan (UNITS = 64 or 128). Block = 64 batch rows, 8 waves.
// Wave w: m-tile w>>1, column groups cg0..cg0+CGW-1; for each cg it computes
// the element-aligned i/f/g/o gate tiles so gates + cell update run on WMMA
// accumulators. c lives in VGPRs for all T steps, h in double-buffered LDS.
// xz_t tiles (contiguous 64 x 4U f16 rows) are prefetched one step ahead
// into dynamic-LDS via GLOBAL_LOAD_ASYNC_TO_LDS_B128, synchronized with
// s_wait_asynccnt + the per-step workgroup barrier.
// ---------------------------------------------------------------------------
template <int UNITS>
__global__ __launch_bounds__(256) void lstm_scan_kernel(
    const _Float16* __restrict__ xz, const _Float16* __restrict__ Upk,
    _Float16* __restrict__ hout, int T, int Bt, int seq)
{
    constexpr int CG   = UNITS / 16;
    constexpr int CGW  = CG / 2;
    constexpr int KT   = UNITS / 32;
    constexpr int N4   = 4 * UNITS;
    constexpr int P    = UNITS + 8;       // h pitch (16B-aligned rows)
    constexpr int TILE = 64 * N4;         // halves per xz step tile

    __shared__ _Float16 h_sh[2][64][P];
    extern __shared__ __align__(16) _Float16 xz_sh[];   // [2][TILE] dynamic

    const int w    = threadIdx.x >> 5;
    const int lane = threadIdx.x & 31;
    const int l15  = lane & 15;
    const int hb   = lane >> 4;
    const int kbase = hb << 3;            // also the M offset for C-layout
    const int mt   = w >> 1;
    const int cg0  = (w & 1) * CGW;
    const int b0   = blockIdx.x * 64;

    // async-stage one step's xz tile (contiguous 64*N4 halves) into buffer
    auto stage = [&](int t, int buf) {
        const char* src = (const char*)(xz + ((long long)t * Bt + b0) * N4);
        unsigned lds0 = (unsigned)(uintptr_t)(xz_sh + (long long)buf * TILE);
        constexpr int CHUNKS = (TILE * 2) / 16;          // 16B copies
        for (int c16 = threadIdx.x; c16 < CHUNKS; c16 += 256) {
            unsigned           ldsa = lds0 + c16 * 16;
            unsigned long long ga   = (unsigned long long)(uintptr_t)src + c16 * 16;
            asm volatile("global_load_async_to_lds_b128 %0, %1, off"
                         :: "v"(ldsa), "v"(ga) : "memory");
        }
    };

    // prologue: stage xz_0, zero h buffer 0 (h_0 = 0), zero c
    stage(0, 0);
    _Float16* hz = &h_sh[0][0][0];
    for (int idx = threadIdx.x; idx < 64 * P; idx += 256) hz[idx] = (_Float16)0.0f;
    v8f c[CGW];
#pragma unroll
    for (int i = 0; i < CGW; ++i)
#pragma unroll
        for (int j = 0; j < 8; ++j) c[i][j] = 0.0f;
    asm volatile("s_wait_asynccnt 0x0" ::: "memory");
    __syncthreads();

    const int row = mt * 16 + l15;
    for (int t = 0; t < T; ++t) {
        const int rd = t & 1, wr = rd ^ 1;
        const _Float16* xzb = xz_sh + (long long)(t & 1) * TILE;

        // kick off async prefetch of next step's xz tile into the other buffer
        if (t + 1 < T) stage(t + 1, (t + 1) & 1);

        // gather A fragments of h from LDS (two 16B ds loads per fragment)
        v16h a[KT];
#pragma unroll
        for (int kt = 0; kt < KT; ++kt) {
            const _Float16* rp = &h_sh[rd][row][kt * 32 + kbase];
            v8h lo = *(const v8h*)rp;
            v8h hi = *(const v8h*)(rp + 16);
#pragma unroll
            for (int j = 0; j < 8; ++j) { a[kt][j] = lo[j]; a[kt][8 + j] = hi[j]; }
        }

#pragma unroll
        for (int i = 0; i < CGW; ++i) {
            const int cg = cg0 + i;
            v8f g4[4];
#pragma unroll
            for (int gidx = 0; gidx < 4; ++gidx) {
                const int nt = gidx * CG + cg;
                const int n  = nt * 16 + l15;
                v8f acc;
#pragma unroll
                for (int vr = 0; vr < 8; ++vr) {
                    int m = mt * 16 + vr + kbase;
                    acc[vr] = (float)xzb[m * N4 + n];
                }
#pragma unroll
                for (int kt = 0; kt < KT; ++kt) {
                    v16h b = *(const v16h*)(Upk +
                        (((long long)kt * (N4 >> 4) + nt) * 32 + lane) * 16);
                    acc = __builtin_amdgcn_wmma_f32_16x16x32_f16(
                        false, a[kt], false, b, (short)0, acc, false, false);
                }
                g4[gidx] = acc;
            }
            // gates + cell update, entirely in registers
#pragma unroll
            for (int vr = 0; vr < 8; ++vr) {
                float ig = 1.0f / (1.0f + __expf(-g4[0][vr]));
                float fg = 1.0f / (1.0f + __expf(-g4[1][vr]));
                float gg = fmaxf(g4[2][vr], 0.0f);     // candidate relu
                float og = 1.0f / (1.0f + __expf(-g4[3][vr]));
                float cv = fg * c[i][vr] + ig * gg;
                c[i][vr] = cv;
                float hv = og * fmaxf(cv, 0.0f);       // output relu
                _Float16 hh = (_Float16)hv;
                int m = mt * 16 + vr + kbase;
                int n = cg * 16 + l15;
                h_sh[wr][m][n] = hh;
                if (seq)
                    hout[((long long)t * Bt + b0 + m) * UNITS + n] = hh;
                else if (t == T - 1)
                    hout[(long long)(b0 + m) * UNITS + n] = hh;
            }
        }
        // this wave's async copies done; barrier publishes h(t) + xz(t+1)
        asm volatile("s_wait_asynccnt 0x0" ::: "memory");
        __syncthreads();
    }
}

// ---------------------------------------------------------------------------
// Dense head: 64->64 relu, 64->32 relu, 32->101, softmax. One thread per row.
// ---------------------------------------------------------------------------
__global__ __launch_bounds__(128) void dense_head_kernel(
    const _Float16* __restrict__ h3,
    const float* __restrict__ Wd1, const float* __restrict__ bd1,
    const float* __restrict__ Wd2, const float* __restrict__ bd2,
    const float* __restrict__ Wd3, const float* __restrict__ bd3,
    float* __restrict__ out, int Bt)
{
    int b = blockIdx.x * 128 + threadIdx.x;
    if (b >= Bt) return;
    float h[64];
    for (int j = 0; j < 64; ++j) h[j] = (float)h3[(long long)b * 64 + j];
    float a1[64];
    for (int j = 0; j < 64; ++j) {
        float s = bd1[j];
        for (int k = 0; k < 64; ++k) s += h[k] * Wd1[k * 64 + j];
        a1[j] = fmaxf(s, 0.0f);
    }
    float a2[32];
    for (int j = 0; j < 32; ++j) {
        float s = bd2[j];
        for (int k = 0; k < 64; ++k) s += a1[k] * Wd2[k * 32 + j];
        a2[j] = fmaxf(s, 0.0f);
    }
    float lg[101];
    float mx = -1e30f;
    for (int j = 0; j < 101; ++j) {
        float s = bd3[j];
        for (int k = 0; k < 32; ++k) s += a2[k] * Wd3[k * 101 + j];
        lg[j] = s;
        mx = fmaxf(mx, s);
    }
    float sum = 0.0f;
    for (int j = 0; j < 101; ++j) { lg[j] = __expf(lg[j] - mx); sum += lg[j]; }
    float inv = 1.0f / sum;
    for (int j = 0; j < 101; ++j) out[(long long)b * 101 + j] = lg[j] * inv;
}

// ---------------------------------------------------------------------------
extern "C" void kernel_launch(void* const* d_in, const int* in_sizes, int n_in,
                              void* d_out, int out_size, void* d_ws, size_t ws_size,
                              hipStream_t stream)
{
    (void)in_sizes; (void)n_in; (void)out_size; (void)ws_size;
    const float* x   = (const float*)d_in[0];
    const float* W1  = (const float*)d_in[1];
    const float* U1  = (const float*)d_in[2];
    const float* b1  = (const float*)d_in[3];
    const float* W2  = (const float*)d_in[4];
    const float* U2  = (const float*)d_in[5];
    const float* b2  = (const float*)d_in[6];
    const float* W3  = (const float*)d_in[7];
    const float* U3  = (const float*)d_in[8];
    const float* b3  = (const float*)d_in[9];
    const float* Wd1 = (const float*)d_in[10];
    const float* bd1 = (const float*)d_in[11];
    const float* Wd2 = (const float*)d_in[12];
    const float* bd2 = (const float*)d_in[13];
    const float* Wd3 = (const float*)d_in[14];
    const float* bd3 = (const float*)d_in[15];
    float* out = (float*)d_out;

    const int T = TSTEPS, B = BATCH;
    const long long M = (long long)T * B;   // 491520

    size_t off = 0;
    auto alloc = [&](size_t bytes) -> char* {
        size_t o = (off + 255) & ~(size_t)255;
        off = o + bytes;
        return (char*)d_ws + o;
    };
    _Float16* W1p = (_Float16*)alloc((size_t)160 * 256 * 2);
    _Float16* U1p = (_Float16*)alloc((size_t)64  * 256 * 2);
    _Float16* W2p = (_Float16*)alloc((size_t)64  * 512 * 2);
    _Float16* U2p = (_Float16*)alloc((size_t)128 * 512 * 2);
    _Float16* W3p = (_Float16*)alloc((size_t)128 * 256 * 2);
    _Float16* U3p = (_Float16*)alloc((size_t)64  * 256 * 2);
    _Float16* xz1 = (_Float16*)alloc((size_t)M * 256 * 2);
    _Float16* h1  = (_Float16*)alloc((size_t)M * 64  * 2);
    _Float16* xz2 = (_Float16*)alloc((size_t)M * 512 * 2);
    _Float16* h2  = (_Float16*)alloc((size_t)M * 128 * 2);
    _Float16* xz3 = (_Float16*)alloc((size_t)M * 256 * 2);
    _Float16* h3  = (_Float16*)alloc((size_t)B * 64  * 2);

    auto packGrid = [](int n) { return dim3((n + 255) / 256); };

    // --- pack weights into fragment-major f16 ---
    pack_weight_kernel<<<packGrid(160 * 256), 256, 0, stream>>>(W1, W1p, 132, 160, 256);
    pack_weight_kernel<<<packGrid(64  * 256), 256, 0, stream>>>(U1, U1p, 64, 64, 256);
    pack_weight_kernel<<<packGrid(64  * 512), 256, 0, stream>>>(W2, W2p, 64, 64, 512);
    pack_weight_kernel<<<packGrid(128 * 512), 256, 0, stream>>>(U2, U2p, 128, 128, 512);
    pack_weight_kernel<<<packGrid(128 * 256), 256, 0, stream>>>(W3, W3p, 128, 128, 256);
    pack_weight_kernel<<<packGrid(64  * 256), 256, 0, stream>>>(U3, U3p, 64, 64, 256);

    const int MB = (int)(M / 64);  // 7680 m-blocks

    // dynamic-LDS bytes for xz double buffers in the scans
    const size_t dyn64  = (size_t)2 * 64 * 256 * 2;   //  64 KB
    const size_t dyn128 = (size_t)2 * 64 * 512 * 2;   // 128 KB

    // --- layer 1: xz1 = permute(x) @ W1 + b1 ; scan ---
    gemm_wmma_kernel<float><<<dim3(MB, 256 / 128), 256, 0, stream>>>(
        x, W1p, b1, xz1, (int)M, 256, D_IN, 160, /*perm=*/1, B, T);
    lstm_scan_kernel<64><<<dim3(B / 64), 256, dyn64, stream>>>(xz1, U1p, h1, T, B, 1);

    // --- layer 2 ---
    gemm_wmma_kernel<_Float16><<<dim3(MB, 512 / 128), 256, 0, stream>>>(
        h1, W2p, b2, xz2, (int)M, 512, 64, 64, 0, 0, 0);
    lstm_scan_kernel<128><<<dim3(B / 64), 256, dyn128, stream>>>(xz2, U2p, h2, T, B, 1);

    // --- layer 3 (last h only) ---
    gemm_wmma_kernel<_Float16><<<dim3(MB, 256 / 128), 256, 0, stream>>>(
        h2, W3p, b3, xz3, (int)M, 256, 128, 128, 0, 0, 0);
    lstm_scan_kernel<64><<<dim3(B / 64), 256, dyn64, stream>>>(xz3, U3p, h3, T, B, 0);

    // --- dense head + softmax ---
    dense_head_kernel<<<dim3((B + 127) / 128), 128, 0, stream>>>(
        h3, Wd1, bd1, Wd2, bd2, Wd3, bd3, out, B);
}
